// Tidev1Decoder_59811714564391
// MI455X (gfx1250) — matrix operand order repaired and checked by reading, not compiled, not run on hardware
//
#include <hip/hip_runtime.h>
#include <hip/hip_bf16.h>
#include <math.h>

// ---------------- problem constants ----------------
#define NQc 900
#define Bc  4
#define Ec  256
#define NHc 8
#define HDc 32
#define DFFc 2048
#define LSc 81
#define LVc 34000
#define NLAYERSc 6
#define TOKc (NQc*Bc)        // 3600
#define TEc  (TOKc*Ec)       // 921600
#define EPSc 1e-5f

typedef __attribute__((ext_vector_type(16))) _Float16 v16h;
typedef __attribute__((ext_vector_type(8)))  _Float16 h8;
typedef __attribute__((ext_vector_type(8)))  float    v8f;

union HF16 { v16h v; _Float16 h[16]; h8 q2[2]; };
union F8   { v8f  v; float    f[8];  };

// f32 -> f16 elementwise conversion (one-time per call for weights/memory)
__global__ void k_cvt(const float* __restrict__ in, _Float16* __restrict__ out, int n)
{
  const int i = blockIdx.x*blockDim.x + threadIdx.x;
  if (i < n) out[i] = (_Float16)in[i];
}

// ---------------------------------------------------------------------------
// Blocked f16 GEMM:  C[M,N] = act(A[M,K] @ W[N,K]^T + bias[N]).
// One wave computes a 32x64 C tile: 2 A-frags x 4 B-frags -> 8
// v_wmma_f32_16x16x32_f16 per 32-deep K chunk. Tail rows are handled by
// CLAMPING the A row pointer (garbage rows only feed C rows that the store
// guard discards), so the inner loop has no branches / zero-fills at all:
// 12 clause-able b128 loads + 8 WMMAs per chunk.
// ---------------------------------------------------------------------------
__global__ __launch_bounds__(32) void k_gemm(
    const _Float16* __restrict__ A, const _Float16* __restrict__ W,
    const float* __restrict__ bias, const unsigned char* __restrict__ rowmask,
    void* __restrict__ Cout, int M, int N, int K, int relu, int out_f16)
{
  const int lane = threadIdx.x;
  const int half = lane >> 4;        // 0 or 1
  const int l16  = lane & 15;
  const int nt = blockIdx.x, mt = blockIdx.y;

  const int rowA0 = mt*32 + l16;
  const int rowA1 = rowA0 + 16;
  const _Float16* arow0 = A + (size_t)(rowA0 < M ? rowA0 : 0) * K;
  const _Float16* arow1 = A + (size_t)(rowA1 < M ? rowA1 : 0) * K;

  const _Float16* wrow[4];
#pragma unroll
  for (int f = 0; f < 4; ++f)
    wrow[f] = W + (size_t)(nt*64 + f*16 + l16) * K;

  F8 acc[8];
#pragma unroll
  for (int i = 0; i < 8; ++i) acc[i].v = (v8f){};

  for (int k0 = 0; k0 < K; k0 += 32) {
    HF16 a0, a1, b[4];
    // A-frag: elems 0..7 -> K=k0+half*8+j ; elems 8..15 -> K=k0+16+half*8+j
    a0.q2[0] = *reinterpret_cast<const h8*>(arow0 + k0 + half*8);
    a0.q2[1] = *reinterpret_cast<const h8*>(arow0 + k0 + 16 + half*8);
    a1.q2[0] = *reinterpret_cast<const h8*>(arow1 + k0 + half*8);
    a1.q2[1] = *reinterpret_cast<const h8*>(arow1 + k0 + 16 + half*8);
    // B-frag: elems 0..7 -> K=k0+half*8+j ; elems 8..15 -> K=k0+16+half*8+j
#pragma unroll
    for (int f = 0; f < 4; ++f) {
      b[f].q2[0] = *reinterpret_cast<const h8*>(wrow[f] + k0 + half*8);
      b[f].q2[1] = *reinterpret_cast<const h8*>(wrow[f] + k0 + 16 + half*8);
    }
#pragma unroll
    for (int f = 0; f < 4; ++f) {
      acc[f].v = __builtin_amdgcn_wmma_f32_16x16x32_f16(
          false, a0.v, false, b[f].v, (short)0, acc[f].v, false, false);
      acc[4+f].v = __builtin_amdgcn_wmma_f32_16x16x32_f16(
          false, a1.v, false, b[f].v, (short)0, acc[4+f].v, false, false);
    }
  }
  // C layout per tile: VGPR r, lane -> row = base + r + 8*half, col = l16
#pragma unroll
  for (int mi = 0; mi < 2; ++mi) {
#pragma unroll
    for (int r = 0; r < 8; ++r) {
      const int row = mt*32 + mi*16 + r + half*8;
      if (row >= M) continue;
      const bool zero = rowmask && rowmask[row];
#pragma unroll
      for (int f = 0; f < 4; ++f) {
        const int coln = nt*64 + f*16 + l16;
        float v = acc[mi*4 + f].f[r] + (bias ? bias[coln] : 0.f);
        if (relu) v = fmaxf(v, 0.f);
        if (zero) v = 0.f;
        const size_t idx = (size_t)row*N + coln;
        if (out_f16) ((_Float16*)Cout)[idx] = (_Float16)v;
        else         ((float*)Cout)[idx]    = v;
      }
    }
  }
}

// ---------------------------------------------------------------------------
// Flash attention over f16 Q/K/V, one wave per (b, h, 16-query tile).
// hd = 32 = one WMMA K-depth. Online softmax; P goes C->A layout through a
// 1 KB LDS bounce. Tail handling is branch-free: Q/K/V row pointers are
// clamped (out-of-range scores are overwritten with -1e30 after the WMMA,
// and out-of-range V rows are weighted by P==0).
// ---------------------------------------------------------------------------
__global__ __launch_bounds__(32) void k_attn(
    const _Float16* __restrict__ Q, const _Float16* __restrict__ K,
    const _Float16* __restrict__ V, const unsigned char* __restrict__ kpm,
    _Float16* __restrict__ O, int Sq, int Sk, int kv_batch_major, float scale)
{
  __shared__ _Float16 plds[16*32];
  const int lane = threadIdx.x;
  const int half = lane >> 4, l16 = lane & 15;
  const int bh = blockIdx.x;
  const int b = bh / NHc, h = bh % NHc;
  const int qt = blockIdx.y;

  // Q A-fragment (16 queries x 32 dims), row clamped
  HF16 aq;
  const int rowq = qt*16 + l16;
  const _Float16* qp = Q + ((size_t)(rowq < Sq ? rowq : 0)*Bc + b)*Ec + h*HDc;
  aq.q2[0] = *reinterpret_cast<const h8*>(qp + half*8);
  aq.q2[1] = *reinterpret_cast<const h8*>(qp + 16 + half*8);

  F8 o0, o1; o0.v = (v8f){}; o1.v = (v8f){};
  float runmax[8], denom[8];
  for (int r = 0; r < 8; ++r) { runmax[r] = -1e30f; denom[r] = 0.f; }

  for (int kc = 0; kc < Sk; kc += 32) {
    // ---- scores: S(16x32) = Q(16x32) @ K^T, two 16-col WMMAs ----
    const int key0 = kc + l16, key1 = kc + 16 + l16;
    const bool ok0 = key0 < Sk, ok1 = key1 < Sk;
    HF16 kb0, kb1;
    {
      const size_t kr0 = kv_batch_major ? ((size_t)b*Sk + (ok0?key0:0))
                                        : ((size_t)(ok0?key0:0)*Bc + b);
      const size_t kr1 = kv_batch_major ? ((size_t)b*Sk + (ok1?key1:0))
                                        : ((size_t)(ok1?key1:0)*Bc + b);
      const _Float16* kp0 = K + kr0*Ec + h*HDc + half*8;
      const _Float16* kp1 = K + kr1*Ec + h*HDc + half*8;
      kb0.q2[0] = *reinterpret_cast<const h8*>(kp0);
      kb0.q2[1] = *reinterpret_cast<const h8*>(kp0 + 16);
      kb1.q2[0] = *reinterpret_cast<const h8*>(kp1);
      kb1.q2[1] = *reinterpret_cast<const h8*>(kp1 + 16);
    }
    F8 c0, c1; c0.v=(v8f){}; c1.v=(v8f){};
    c0.v = __builtin_amdgcn_wmma_f32_16x16x32_f16(false, aq.v, false, kb0.v,(short)0, c0.v,false,false);
    c1.v = __builtin_amdgcn_wmma_f32_16x16x32_f16(false, aq.v, false, kb1.v,(short)0, c1.v,false,false);

    bool m0 = !ok0, m1 = !ok1;
    if (kpm) {
      if (!m0) m0 = (kpm[(size_t)b*Sk + key0] != 0);
      if (!m1) m1 = (kpm[(size_t)b*Sk + key1] != 0);
    }
    // ---- online softmax per row (16 lanes of a half share a row) ----
    for (int r = 0; r < 8; ++r) {
      float s0 = m0 ? -1e30f : c0.f[r]*scale;
      float s1 = m1 ? -1e30f : c1.f[r]*scale;
      float t = fmaxf(s0, s1);
      for (int m = 1; m < 16; m <<= 1) t = fmaxf(t, __shfl_xor(t, m, 32));
      const float newm = fmaxf(runmax[r], t);
      const float corr = __expf(runmax[r] - newm);
      const float p0 = __expf(s0 - newm);
      const float p1 = __expf(s1 - newm);
      float rs = p0 + p1;
      for (int m = 1; m < 16; m <<= 1) rs += __shfl_xor(rs, m, 32);
      denom[r] = denom[r]*corr + rs;
      runmax[r] = newm;
      o0.f[r] *= corr; o1.f[r] *= corr;
      const int prow = r + half*8;
      plds[prow*32 + l16]      = (_Float16)p0;
      plds[prow*32 + 16 + l16] = (_Float16)p1;
    }
    __syncthreads();
    // ---- P (16x32) as A-fragment ----
    HF16 pa;
    for (int j = 0; j < 8; ++j) {
      pa.h[j]   = plds[l16*32 + half*16 + j];
      pa.h[8+j] = plds[l16*32 + half*16 + 8 + j];
    }
    // ---- V chunk B-fragments: contraction = key-in-chunk, col = head dim ----
    HF16 vb0, vb1;
    for (int j = 0; j < 16; ++j) {
      const int kk = kc + half*16 + j;
      const size_t vr = kv_batch_major ? ((size_t)b*Sk + (kk < Sk ? kk : 0))
                                       : ((size_t)(kk < Sk ? kk : 0)*Bc + b);
      const _Float16* vp = V + vr*Ec + h*HDc;
      vb0.h[j] = vp[l16];
      vb1.h[j] = vp[16 + l16];
    }
    o0.v = __builtin_amdgcn_wmma_f32_16x16x32_f16(false, pa.v, false, vb0.v,(short)0, o0.v,false,false);
    o1.v = __builtin_amdgcn_wmma_f32_16x16x32_f16(false, pa.v, false, vb1.v,(short)0, o1.v,false,false);
    __syncthreads();
  }
  // ---- write O, f16, token-major rows q*B+b ----
  for (int r = 0; r < 8; ++r) {
    const int row = qt*16 + r + half*8;
    if (row < Sq) {
      const float inv = 1.f / denom[r];
      const size_t base = ((size_t)row*Bc + b)*Ec + h*HDc + l16;
      O[base]      = (_Float16)(o0.f[r]*inv);
      O[base + 16] = (_Float16)(o1.f[r]*inv);
    }
  }
}

// ---------------------------------------------------------------------------
// LayerNorm over E=256: out = LN(x (+res)) * s + b.  One wave per row.
// Optionally also writes an f16 copy (for downstream GEMM A-inputs).
// ---------------------------------------------------------------------------
__global__ __launch_bounds__(32) void k_ln(
    const float* __restrict__ x, const float* __restrict__ res,
    const float* __restrict__ s, const float* __restrict__ bsh,
    float* __restrict__ out, _Float16* __restrict__ out16)
{
  const int row = blockIdx.x, lane = threadIdx.x;
  const size_t base = (size_t)row*Ec + lane*8;
  float v[8];
  float sum = 0.f;
  for (int i = 0; i < 8; ++i) {
    float t = x[base + i];
    if (res) t += res[base + i];
    v[i] = t; sum += t;
  }
  for (int m = 1; m < 32; m <<= 1) sum += __shfl_xor(sum, m, 32);
  const float mean = sum * (1.0f/Ec);
  float sq = 0.f;
  for (int i = 0; i < 8; ++i) { const float d = v[i]-mean; sq += d*d; }
  for (int m = 1; m < 32; m <<= 1) sq += __shfl_xor(sq, m, 32);
  const float rinv = rsqrtf(sq*(1.0f/Ec) + EPSc);
  for (int i = 0; i < 8; ++i) {
    const int col = lane*8 + i;
    const float y = (v[i]-mean)*rinv*s[col] + bsh[col];
    out[(size_t)row*Ec + col] = y;
    if (out16) out16[(size_t)row*Ec + col] = (_Float16)y;
  }
}

// z(f16) = a + b  (query+pos; consumed only by GEMMs)
__global__ void k_add_h(const float* __restrict__ a, const float* __restrict__ b,
                        _Float16* __restrict__ c, int n)
{
  const int i = blockIdx.x*blockDim.x + threadIdx.x;
  if (i < n) c[i] = (_Float16)(a[i] + b[i]);
}

// softmax over groups of 16 (attention weights per (token, head))
__global__ void k_awsm(float* __restrict__ aw, int ngroups)
{
  const int g = blockIdx.x*blockDim.x + threadIdx.x;
  if (g >= ngroups) return;
  float* p = aw + (size_t)g*16;
  float mx = p[0];
  for (int i = 1; i < 16; ++i) mx = fmaxf(mx, p[i]);
  float sum = 0.f; float e[16];
  for (int i = 0; i < 16; ++i) { e[i] = __expf(p[i]-mx); sum += e[i]; }
  const float inv = 1.f/sum;
  for (int i = 0; i < 16; ++i) p[i] = e[i]*inv;
}

__device__ inline float dsamp(const _Float16* __restrict__ vd, int b, int st,
                              int H, int W, int xi, int yi, int col)
{
  if (xi < 0 || xi >= W || yi < 0 || yi >= H) return 0.f;
  const size_t idx = ((size_t)b*LVc + st + (size_t)yi*W + xi)*Ec + col;
  return (float)vd[idx];
}

// deformable-attention bilinear sampling; thread per (token, head, dim)
__global__ void k_sample(const _Float16* __restrict__ vdef,
                         const float* __restrict__ off,
                         const float* __restrict__ aw,
                         const float* __restrict__ ref,
                         _Float16* __restrict__ out)
{
  const int tid = blockIdx.x*blockDim.x + threadIdx.x;
  if (tid >= TOKc*NHc*HDc) return;
  const int d = tid & 31;
  const int h = (tid >> 5) & 7;
  const int t = tid >> 8;          // token = q*B + b
  const int b = t & (Bc-1);
  const int Hs[4] = {160, 80, 40, 20};
  const int Ws[4] = {160, 80, 40, 20};
  const int st[4] = {0, 25600, 32000, 33600};
  const int col = h*HDc + d;
  float acc = 0.f;
  for (int l = 0; l < 4; ++l) {
    const float rx = ref[((size_t)t*4 + l)*2 + 0];
    const float ry = ref[((size_t)t*4 + l)*2 + 1];
    const float Wf = (float)Ws[l], Hf = (float)Hs[l];
    for (int p = 0; p < 4; ++p) {
      const size_t ob = (size_t)t*256 + (size_t)((h*4 + l)*4 + p)*2;
      const float ox = off[ob], oy = off[ob + 1];
      const float a  = aw[(size_t)t*128 + h*16 + l*4 + p];
      const float x = (rx + ox/Wf)*Wf - 0.5f;
      const float y = (ry + oy/Hf)*Hf - 0.5f;
      const float x0 = floorf(x), y0 = floorf(y);
      const float wx = x - x0, wy = y - y0;
      const int xi = (int)x0, yi = (int)y0;
      const float s00 = dsamp(vdef,b,st[l],Hs[l],Ws[l],xi,  yi,  col);
      const float s10 = dsamp(vdef,b,st[l],Hs[l],Ws[l],xi+1,yi,  col);
      const float s01 = dsamp(vdef,b,st[l],Hs[l],Ws[l],xi,  yi+1,col);
      const float s11 = dsamp(vdef,b,st[l],Hs[l],Ws[l],xi+1,yi+1,col);
      acc += a * (s00*(1.f-wx)*(1.f-wy) + s10*wx*(1.f-wy)
                + s01*(1.f-wx)*wy       + s11*wx*wy);
    }
  }
  out[(size_t)t*Ec + col] = (_Float16)acc;
}

// ---------------------------------------------------------------------------
extern "C" void kernel_launch(void* const* d_in, const int* in_sizes, int n_in,
                              void* d_out, int out_size, void* d_ws, size_t ws_size,
                              hipStream_t stream)
{
  (void)in_sizes; (void)n_in; (void)out_size; (void)ws_size;
  const float* tgt   = (const float*)d_in[0];
  const float* pos   = (const float*)d_in[1];
  const float* ref   = (const float*)d_in[2];
  const float* memry = (const float*)d_in[3];
  const float* msupp = (const float*)d_in[4];
  const float* sa_w  = (const float*)d_in[5];
  const float* sa_b  = (const float*)d_in[6];
  const float* sa_ow = (const float*)d_in[7];
  const float* sa_ob = (const float*)d_in[8];
  const float* cs_w  = (const float*)d_in[9];
  const float* cs_b  = (const float*)d_in[10];
  const float* cs_ow = (const float*)d_in[11];
  const float* cs_ob = (const float*)d_in[12];
  const float* n2_s  = (const float*)d_in[13];
  const float* n2_b  = (const float*)d_in[14];
  const float* ncs_s = (const float*)d_in[15];
  const float* ncs_b = (const float*)d_in[16];
  const float* n1_s  = (const float*)d_in[17];
  const float* n1_b  = (const float*)d_in[18];
  const float* n3_s  = (const float*)d_in[19];
  const float* n3_b  = (const float*)d_in[20];
  const float* da_vw = (const float*)d_in[21];
  const float* da_vb = (const float*)d_in[22];
  const float* da_offw = (const float*)d_in[23];
  const float* da_offb = (const float*)d_in[24];
  const float* da_aww  = (const float*)d_in[25];
  const float* da_awb  = (const float*)d_in[26];
  const float* da_ow = (const float*)d_in[27];
  const float* da_ob = (const float*)d_in[28];
  const float* ff1_w = (const float*)d_in[29];
  const float* ff1_b = (const float*)d_in[30];
  const float* ff2_w = (const float*)d_in[31];
  const float* ff2_b = (const float*)d_in[32];
  const float* fn_s  = (const float*)d_in[33];
  const float* fn_b  = (const float*)d_in[34];
  const unsigned char* smask = (const unsigned char*)d_in[35];
  const unsigned char* vmask = (const unsigned char*)d_in[36];

  // ---------------- workspace arena ----------------
  char* wp = (char*)d_ws;
  auto carve = [&](size_t bytes) -> char* {
    char* p = wp; wp += (bytes + 255) & ~(size_t)255; return p;
  };
  // f32 activations
  float* xbuf = (float*)carve((size_t)TEc*4);
  float* t0   = (float*)carve((size_t)TEc*4);   // sampling offsets (f32)
  float* t1   = (float*)carve((size_t)TEc*4);   // attention weights (f32)
  float* t4   = (float*)carve((size_t)TEc*4);   // pre-residual projection out
  // f16 activations
  _Float16* xbuf16 = (_Float16*)carve((size_t)TEc*2);
  _Float16* qbuf16 = (_Float16*)carve((size_t)TEc*2);
  _Float16* t0h    = (_Float16*)carve((size_t)TEc*2);   // Q
  _Float16* t1h    = (_Float16*)carve((size_t)TEc*2);   // K
  _Float16* t2h    = (_Float16*)carve((size_t)TEc*2);   // V / sample out
  _Float16* t3h    = (_Float16*)carve((size_t)TEc*2);   // attention out
  _Float16* hbuf16 = (_Float16*)carve((size_t)TOKc*DFFc*2);
  _Float16* vdef   = (_Float16*)carve((size_t)Bc*LVc*Ec*2);
  // f16 copies of weights / static inputs (converted once per call)
  _Float16* memry16 = (_Float16*)carve((size_t)Bc*LVc*Ec*2);
  _Float16* msupp16 = (_Float16*)carve((size_t)Bc*LSc*Ec*2);
  const int WSA  = NLAYERSc*3*Ec*Ec;   // 1179648
  const int WEE  = NLAYERSc*Ec*Ec;     // 393216
  const int WAW  = NLAYERSc*128*Ec;    // 196608
  const int WFF  = NLAYERSc*DFFc*Ec;   // 3145728
  _Float16* sa_w16   = (_Float16*)carve((size_t)WSA*2);
  _Float16* sa_ow16  = (_Float16*)carve((size_t)WEE*2);
  _Float16* cs_w16   = (_Float16*)carve((size_t)WSA*2);
  _Float16* cs_ow16  = (_Float16*)carve((size_t)WEE*2);
  _Float16* da_vw16  = (_Float16*)carve((size_t)WEE*2);
  _Float16* da_offw16= (_Float16*)carve((size_t)WEE*2);
  _Float16* da_aww16 = (_Float16*)carve((size_t)WAW*2);
  _Float16* da_ow16  = (_Float16*)carve((size_t)WEE*2);
  _Float16* ff1_w16  = (_Float16*)carve((size_t)WFF*2);
  _Float16* ff2_w16  = (_Float16*)carve((size_t)WFF*2);

  auto cvt = [&](const float* src, _Float16* dst, size_t n) {
    k_cvt<<<(int)((n + 255)/256), 256, 0, stream>>>(src, dst, (int)n);
  };
  // one-time conversions
  cvt(memry, memry16, (size_t)Bc*LVc*Ec);
  cvt(msupp, msupp16, (size_t)Bc*LSc*Ec);
  cvt(sa_w,  sa_w16,  WSA);   cvt(sa_ow, sa_ow16, WEE);
  cvt(cs_w,  cs_w16,  WSA);   cvt(cs_ow, cs_ow16, WEE);
  cvt(da_vw, da_vw16, WEE);   cvt(da_offw, da_offw16, WEE);
  cvt(da_aww,da_aww16,WAW);   cvt(da_ow, da_ow16, WEE);
  cvt(ff1_w, ff1_w16, WFF);   cvt(ff2_w, ff2_w16, WFF);
  cvt(tgt,   xbuf16,  TEc);   // x(f16) before first LN refresh

  hipMemcpyAsync(xbuf, tgt, (size_t)TEc*4, hipMemcpyDeviceToDevice, stream);

  auto gemm = [&](const _Float16* A, const _Float16* W, const float* bias, void* C,
                  int M, int N, int K, int relu, int of16,
                  const unsigned char* rmask) {
    dim3 g(N/64, (M+31)/32);           // 32x64 C tile per wave
    k_gemm<<<g, dim3(32), 0, stream>>>(A, W, bias, rmask, C, M, N, K, relu, of16);
  };
  const float scale = 0.17677669529663687f;    // 1/sqrt(32)
  const int qtiles = (NQc + 15)/16;            // 57

  for (int l = 0; l < NLAYERSc; ++l) {
    // ---------------- self-attention ----------------
    const _Float16* wq = sa_w16 + (size_t)l*3*Ec*Ec;
    const _Float16* wk = wq + (size_t)Ec*Ec;
    const _Float16* wv = wk + (size_t)Ec*Ec;
    const float* bq = sa_b + (size_t)l*3*Ec;
    const float* bk = bq + Ec; const float* bv = bk + Ec;
    k_add_h<<<(TEc+255)/256, 256, 0, stream>>>(xbuf, pos, qbuf16, TEc);
    gemm(qbuf16, wq, bq, t0h, TOKc, Ec, Ec, 0, 1, nullptr);
    gemm(qbuf16, wk, bk, t1h, TOKc, Ec, Ec, 0, 1, nullptr);
    gemm(xbuf16, wv, bv, t2h, TOKc, Ec, Ec, 0, 1, nullptr);
    k_attn<<<dim3(Bc*NHc, qtiles), 32, 0, stream>>>(t0h, t1h, t2h, nullptr, t3h,
                                                    NQc, NQc, 0, scale);
    gemm(t3h, sa_ow16 + (size_t)l*Ec*Ec, sa_ob + (size_t)l*Ec, t4, TOKc, Ec, Ec, 0, 0, nullptr);
    k_ln<<<TOKc, 32, 0, stream>>>(xbuf, t4, n2_s + (size_t)l*Ec, n2_b + (size_t)l*Ec,
                                  xbuf, xbuf16);

    // ---------------- cross-attention vs support ----------------
    const _Float16* cwq = cs_w16 + (size_t)l*3*Ec*Ec;
    const _Float16* cwk = cwq + (size_t)Ec*Ec;
    const _Float16* cwv = cwk + (size_t)Ec*Ec;
    const float* cbq = cs_b + (size_t)l*3*Ec;
    const float* cbk = cbq + Ec; const float* cbv = cbk + Ec;
    k_add_h<<<(TEc+255)/256, 256, 0, stream>>>(xbuf, pos, qbuf16, TEc);
    gemm(qbuf16,  cwq, cbq, t0h, TOKc,   Ec, Ec, 0, 1, nullptr);
    gemm(msupp16, cwk, cbk, t1h, Bc*LSc, Ec, Ec, 0, 1, nullptr);   // rows b*LS+s
    gemm(msupp16, cwv, cbv, t2h, Bc*LSc, Ec, Ec, 0, 1, nullptr);
    k_attn<<<dim3(Bc*NHc, qtiles), 32, 0, stream>>>(t0h, t1h, t2h, smask, t3h,
                                                    NQc, LSc, 1, scale);
    gemm(t3h, cs_ow16 + (size_t)l*Ec*Ec, cs_ob + (size_t)l*Ec, t4, TOKc, Ec, Ec, 0, 0, nullptr);
    k_ln<<<TOKc, 32, 0, stream>>>(xbuf, t4, ncs_s + (size_t)l*Ec, ncs_b + (size_t)l*Ec,
                                  xbuf, xbuf16);

    // ---------------- deformable attention ----------------
    k_add_h<<<(TEc+255)/256, 256, 0, stream>>>(xbuf, pos, qbuf16, TEc);
    gemm(memry16, da_vw16 + (size_t)l*Ec*Ec, da_vb + (size_t)l*Ec, vdef,
         Bc*LVc, Ec, Ec, 0, 1, vmask);                          // f16 value tensor
    gemm(qbuf16, da_offw16 + (size_t)l*256*Ec, da_offb + (size_t)l*256, t0,
         TOKc, 256, Ec, 0, 0, nullptr);                          // sampling offsets
    gemm(qbuf16, da_aww16 + (size_t)l*128*Ec, da_awb + (size_t)l*128, t1,
         TOKc, 128, Ec, 0, 0, nullptr);                          // attention weights
    k_awsm<<<(TOKc*NHc + 255)/256, 256, 0, stream>>>(t1, TOKc*NHc);
    k_sample<<<(TOKc*NHc*HDc + 255)/256, 256, 0, stream>>>(vdef, t0, t1, ref, t2h);
    gemm(t2h, da_ow16 + (size_t)l*Ec*Ec, da_ob + (size_t)l*Ec, t4, TOKc, Ec, Ec, 0, 0, nullptr);
    k_ln<<<TOKc, 32, 0, stream>>>(xbuf, t4, n1_s + (size_t)l*Ec, n1_b + (size_t)l*Ec,
                                  xbuf, xbuf16);

    // ---------------- FFN ----------------
    gemm(xbuf16, ff1_w16 + (size_t)l*DFFc*Ec, ff1_b + (size_t)l*DFFc, hbuf16,
         TOKc, DFFc, Ec, 1, 1, nullptr);
    gemm(hbuf16, ff2_w16 + (size_t)l*Ec*DFFc, ff2_b + (size_t)l*Ec, t4,
         TOKc, Ec, DFFc, 0, 0, nullptr);
    k_ln<<<TOKc, 32, 0, stream>>>(xbuf, t4, n3_s + (size_t)l*Ec, n3_b + (size_t)l*Ec,
                                  xbuf, xbuf16);
  }

  // final LN -> d_out
  k_ln<<<TOKc, 32, 0, stream>>>(xbuf, nullptr, fn_s, fn_b, (float*)d_out, nullptr);
}
